// ESNReservoir_62191126446136
// MI455X (gfx1250) — matrix operand order repaired and checked by reading, not compiled, not run on hardware
//
#include <hip/hip_runtime.h>
#include <hip/hip_bf16.h>
#include <math.h>

#define R 2048
#define BS 16
#define TT 2048
#define NWG 32          // persistent workgroups
#define WPB 4           // waves per workgroup (wave32)
#define LROW (R + 8)    // padded bf16 elements per LDS W-row (bank stagger)

// LDS layout (bytes)
#define W_BYTES   (WPB * 16 * LROW * 2)            // 4 waves x 16 rows x padded K, bf16
#define CT_OFF    (W_BYTES)                        // per-wave 16x16 f32 C tile
#define WO_OFF    (CT_OFF + WPB * 256 * 4)         // per-wave 3x16 Wout staging
#define SMEM_BYTES (WO_OFF + WPB * 48 * 4)

typedef __attribute__((ext_vector_type(16))) __bf16 v16bf;
typedef __attribute__((ext_vector_type(8)))  float  v8f;

union Frag32B { uint4 u[2]; v16bf v; };

__device__ __forceinline__ unsigned short f32_to_bf16(float f) {
    unsigned int u = __builtin_bit_cast(unsigned int, f);
    unsigned int r = (u + 0x7FFFu + ((u >> 16) & 1u)) >> 16;  // RNE
    return (unsigned short)r;
}

// ---- prep: W fp32 -> bf16, zero both h buffers, zero the grid-barrier counter
__global__ void esn_prep(const float* __restrict__ W, unsigned int* __restrict__ counter,
                         unsigned short* __restrict__ hbuf, unsigned short* __restrict__ Wbf) {
    const size_t idx = (size_t)blockIdx.x * blockDim.x + threadIdx.x;
    const size_t stride = (size_t)gridDim.x * blockDim.x;
    for (size_t i = idx; i < (size_t)R * R; i += stride)
        Wbf[i] = f32_to_bf16(W[i]);
    if (idx < (size_t)2 * BS * R) hbuf[idx] = 0;
    if (idx == 0) *counter = 0u;
}

// ---- prep: out[b,t,o] = bias[o]
__global__ void esn_out_init(float* __restrict__ out, const float* __restrict__ bias) {
    const int idx = blockIdx.x * blockDim.x + threadIdx.x;
    if (idx < BS * TT * 3) out[idx] = bias[idx % 3];
}

__device__ __forceinline__ v8f wmma_step(const unsigned short* hin,
                                         const unsigned short* wlds,
                                         int m, int n, int half, int k0, v8f acc) {
    Frag32B a, b;
    // A = h_prev[16 x 32] : lane holds row m, K = k0 + half*8 + {0..7, 16..23}
    a.u[0] = *(const uint4*)(hin + (size_t)m * R + k0 + half * 8);
    a.u[1] = *(const uint4*)(hin + (size_t)m * R + k0 + 16 + half * 8);
    // B = W^T[32 x 16] : lane holds col n, K = k0 + half*8 + {0..7, 16..23},
    // i.e. bf16 of W row (base_col+n) -- no transpose needed.
    b.u[0] = *(const uint4*)(wlds + n * LROW + k0 + half * 8);
    b.u[1] = *(const uint4*)(wlds + n * LROW + k0 + 16 + half * 8);
    return __builtin_amdgcn_wmma_f32_16x16x32_bf16(
        false, a.v, false, b.v, (short)0, acc, false, false);
}

// ---- persistent ESN recurrence: W resident in LDS, bf16 WMMA, grid barrier per step
extern "C" __global__ __launch_bounds__(128, 1)
void esn_persistent(const float* __restrict__ x,        // [BS, TT, 3]
                    const float* __restrict__ Win,      // [R, 3]
                    const float* __restrict__ Woutw,    // [3, R]
                    const unsigned short* __restrict__ Wbf,  // [R, R] bf16 bits
                    unsigned short* __restrict__ hbuf,  // [2, BS, R] bf16 bits
                    unsigned int* __restrict__ counter,
                    float* __restrict__ out) {          // [BS, TT, 3]
    extern __shared__ __align__(16) unsigned char smem[];

    const int wave = threadIdx.x >> 5;
    const int lane = threadIdx.x & 31;
    const int n    = lane & 15;        // output column within tile / batch row for readout
    const int half = lane >> 4;
    const int base_col = (blockIdx.x * WPB + wave) * 16;
    const int j = base_col + n;        // this lane's reservoir column

    unsigned short* wlds = (unsigned short*)smem + (size_t)wave * 16 * LROW;
    float* ctile = (float*)(smem + CT_OFF) + wave * 256;
    float* wo    = (float*)(smem + WO_OFF) + wave * 48;

    // ---- one-time stage: 16 rows of W (bf16) for this wave into LDS
    for (int r = 0; r < 16; ++r) {
        const unsigned short* src = Wbf + (size_t)(base_col + r) * R;
        unsigned short* dst = wlds + r * LROW;
        for (int k = lane * 8; k < R; k += 32 * 8)
            *(uint4*)(dst + k) = *(const uint4*)(src + k);
    }
    const float win0 = Win[j * 3 + 0];
    const float win1 = Win[j * 3 + 1];
    const float win2 = Win[j * 3 + 2];
    if (lane < 16) {
        wo[0 * 16 + n] = Woutw[0 * R + j];
        wo[1 * 16 + n] = Woutw[1 * R + j];
        wo[2 * 16 + n] = Woutw[2 * R + j];
    }
    __syncthreads();

    const int m = n;  // A-fragment row (batch index) this lane loads

    for (int t = 0; t < TT; ++t) {
        const unsigned short* hin  = hbuf + (size_t)(t & 1) * BS * R;
        unsigned short*       hout = hbuf + (size_t)((t + 1) & 1) * BS * R;

        // 4 independent accumulator chains -> 4x WMMA ILP (TRANS co-exec)
        v8f acc0 = {}, acc1 = {}, acc2 = {}, acc3 = {};
        for (int kk = 0; kk < R / 32; kk += 4) {
            __builtin_prefetch(hin + (size_t)m * R + kk * 32 + 128, 0, 1);
            acc0 = wmma_step(hin, wlds, m, n, half, (kk + 0) * 32, acc0);
            acc1 = wmma_step(hin, wlds, m, n, half, (kk + 1) * 32, acc1);
            acc2 = wmma_step(hin, wlds, m, n, half, (kk + 2) * 32, acc2);
            acc3 = wmma_step(hin, wlds, m, n, half, (kk + 3) * 32, acc3);
        }
        v8f acc = (acc0 + acc1) + (acc2 + acc3);

        // input term + tanh; C layout: VGPR r -> row (r + 8*half), col n
        #pragma unroll
        for (int r = 0; r < 8; ++r) {
            const int brow = r + 8 * half;
            const float x0 = x[((size_t)brow * TT + t) * 3 + 0];
            const float x1 = x[((size_t)brow * TT + t) * 3 + 1];
            const float x2 = x[((size_t)brow * TT + t) * 3 + 2];
            float v = acc[r] + x0 * win0 + x1 * win1 + x2 * win2;
            float h = tanhf(v);
            hout[(size_t)brow * R + j] = f32_to_bf16(h);
            ctile[brow * 16 + n] = h;
        }
        __syncthreads();

        // readout partials for this wave's 16 columns: lanes 0..15 own batch row b
        if (lane < 16) {
            const int b = lane;
            float s0 = 0.f, s1 = 0.f, s2 = 0.f;
            #pragma unroll
            for (int q = 0; q < 16; ++q) {
                const float hv = ctile[b * 16 + q];
                s0 += hv * wo[0 * 16 + q];
                s1 += hv * wo[1 * 16 + q];
                s2 += hv * wo[2 * 16 + q];
            }
            float* op = out + ((size_t)b * TT + t) * 3;
            atomicAdd(op + 0, s0);
            atomicAdd(op + 1, s1);
            atomicAdd(op + 2, s2);
        }

        // ---- grid-wide barrier: h_t fully published before any WG starts t+1
        __threadfence();
        __syncthreads();
        if (threadIdx.x == 0) {
            __hip_atomic_fetch_add(counter, 1u, __ATOMIC_RELEASE, __HIP_MEMORY_SCOPE_AGENT);
            const unsigned int target = (unsigned int)(t + 1) * NWG;
            while (__hip_atomic_load(counter, __ATOMIC_ACQUIRE, __HIP_MEMORY_SCOPE_AGENT) < target)
                __builtin_amdgcn_s_sleep(4);
        }
        __syncthreads();
        __threadfence();
    }
}

extern "C" void kernel_launch(void* const* d_in, const int* in_sizes, int n_in,
                              void* d_out, int out_size, void* d_ws, size_t ws_size,
                              hipStream_t stream) {
    const float* x     = (const float*)d_in[0];   // [16, 2048, 3]
    const float* Win   = (const float*)d_in[1];   // [2048, 3]
    const float* W     = (const float*)d_in[2];   // [2048, 2048]
    const float* Woutw = (const float*)d_in[3];   // [3, 2048]
    const float* Woutb = (const float*)d_in[4];   // [3]
    float* out = (float*)d_out;                   // [16, 2048, 3]

    unsigned char* ws = (unsigned char*)d_ws;
    unsigned int*   counter = (unsigned int*)ws;                       // 256 B slot
    unsigned short* hbuf    = (unsigned short*)(ws + 256);             // 2*16*2048*2 = 128 KB
    unsigned short* Wbf     = (unsigned short*)(ws + 256 + 131072);    // 8 MB bf16 W

    esn_prep<<<2048, 256, 0, stream>>>(W, counter, hbuf, Wbf);
    esn_out_init<<<(BS * TT * 3 + 255) / 256, 256, 0, stream>>>(out, Woutb);
    esn_persistent<<<NWG, 128, SMEM_BYTES, stream>>>(x, Win, Woutw, Wbf, hbuf, counter, out);
}
